// TransformerBlock_86681029968615
// MI455X (gfx1250) — compile-verified, matrix operand order checked by reading, not run on hardware
//
#include <hip/hip_runtime.h>
#include <hip/hip_bf16.h>

typedef __attribute__((ext_vector_type(16))) __bf16 v16bf;
typedef __attribute__((ext_vector_type(8)))  float  v8f;
typedef unsigned short ushort_t;

#define D_MODEL 768
#define N_HEADS 12
#define D_FF    3072
#define DK      64
#define BATCH   2
#define SEQ     2048
#define MS      (BATCH * SEQ)   // 4096 total rows

// Tensor Data Mover availability (device pass only; host pass takes fallback)
#if defined(__AMDGCN__) && __has_builtin(__builtin_amdgcn_tensor_load_to_lds) && \
    __has_builtin(__builtin_amdgcn_s_wait_tensorcnt)
#define HAS_TDM 1
#else
#define HAS_TDM 0
#endif

// VALU lane shuffle (v_permlane16_b32) for softmax butterfly reductions
#if defined(__AMDGCN__) && __has_builtin(__builtin_amdgcn_permlane16)
#define HAS_PERMLANE 1
#else
#define HAS_PERMLANE 0
#endif

union FragBF {
    v16bf    v;
    uint4    q[2];
    ushort_t s[16];
};

__device__ __forceinline__ ushort_t f2bf(float f) {
    unsigned u = __builtin_bit_cast(unsigned, f);
    unsigned r = u + 0x7FFFu + ((u >> 16) & 1u);   // round-to-nearest-even
    return (ushort_t)(r >> 16);
}

#if HAS_PERMLANE
__device__ __forceinline__ float permxor16(float v, int lo, int hi) {
    int x = __builtin_bit_cast(int, v);
    int r = __builtin_amdgcn_permlane16(x, x, lo, hi, false, false);
    return __builtin_bit_cast(float, r);
}
#endif

// max-reduce across the 16 lanes of each half-wave (pure VALU when available)
__device__ __forceinline__ float redmax16(float v) {
#if HAS_PERMLANE
    v = fmaxf(v, permxor16(v, (int)0x67452301u, (int)0xEFCDAB89u)); // xor 1
    v = fmaxf(v, permxor16(v, (int)0x54761032u, (int)0xDCFE98BAu)); // xor 2
    v = fmaxf(v, permxor16(v, (int)0x32107654u, (int)0xBA98FEDCu)); // xor 4
    v = fmaxf(v, permxor16(v, (int)0xFEDCBA98u, (int)0x76543210u)); // xor 8
#else
#pragma unroll
    for (int m = 1; m < 16; m <<= 1)
        v = fmaxf(v, __shfl_xor(v, m, 32));
#endif
    return v;
}

// sum-reduce across the 16 lanes of each half-wave
__device__ __forceinline__ float redsum16(float v) {
#if HAS_PERMLANE
    v += permxor16(v, (int)0x67452301u, (int)0xEFCDAB89u);
    v += permxor16(v, (int)0x54761032u, (int)0xDCFE98BAu);
    v += permxor16(v, (int)0x32107654u, (int)0xBA98FEDCu);
    v += permxor16(v, (int)0xFEDCBA98u, (int)0x76543210u);
#else
#pragma unroll
    for (int m = 1; m < 16; m <<= 1)
        v += __shfl_xor(v, m, 32);
#endif
    return v;
}

#if HAS_TDM
typedef unsigned int u32x4 __attribute__((ext_vector_type(4)));
typedef int          i32x4 __attribute__((ext_vector_type(4)));
typedef int          i32x8 __attribute__((ext_vector_type(8)));

// Issue one 2D TDM tile load: tile_w x tile_h of bf16 elements, global row
// stride = row_stride elements, packed row-major into LDS at ldst.
// Per ISA cdna5 §8: D# group0 = {flags, lds_addr, global_addr, type=2},
// group1 = {data_size, tensor dims, tile dims, dim0 stride}. tensor==tile so
// no OOB clamping is exercised. Descriptor values are wave-uniform.
__device__ __forceinline__ void tdm_load_2d(const void* gsrc, void* ldst,
                                            unsigned tile_w, unsigned tile_h,
                                            unsigned row_stride) {
    unsigned long long ga = (unsigned long long)gsrc;
    unsigned la = (unsigned)(unsigned long long)ldst;   // LDS byte offset
    u32x4 g0;
    g0[0] = 1u;                                          // count=1, user D#
    g0[1] = la;                                          // lds_addr
    g0[2] = (unsigned)(ga & 0xFFFFFFFFull);              // global_addr[31:0]
    g0[3] = (unsigned)((ga >> 32) & 0x1FFFFFFull) | (2u << 30); // [56:32]|type=2
    i32x8 g1;
    g1[0] = (int)(1u << 16);                             // data_size=1 (2 bytes)
    g1[1] = (int)((tile_w & 0xFFFFu) << 16);             // tensor_dim0[15:0]
    g1[2] = (int)(((tile_w >> 16) & 0xFFFFu) | ((tile_h & 0xFFFFu) << 16)); // dim0 hi | dim1 lo
    g1[3] = (int)(((tile_h >> 16) & 0xFFFFu) | ((tile_w & 0xFFFFu) << 16)); // dim1 hi | tile_dim0
    g1[4] = (int)(tile_h & 0xFFFFu);                     // tile_dim1 (tile_dim2=0)
    g1[5] = (int)row_stride;                             // tensor_dim0_stride[31:0]
    g1[6] = 0;
    g1[7] = 0;
    i32x4 gz = {0, 0, 0, 0};
#if __clang_major__ >= 23
    i32x8 gz8 = {0, 0, 0, 0, 0, 0, 0, 0};
    __builtin_amdgcn_tensor_load_to_lds(g0, g1, gz, gz, gz8, 0);
#else
    __builtin_amdgcn_tensor_load_to_lds(g0, g1, gz, gz, 0);
#endif
}
#endif // HAS_TDM

// ---------------------------------------------------------------------------
// Weight convert + transpose: W (K x N, f32, input-major) -> Wt (N x K, bf16)
// ---------------------------------------------------------------------------
__global__ void k_transpose_bf16(const float* __restrict__ src,
                                 ushort_t* __restrict__ dst, int K, int N) {
    int i = blockIdx.x * 256 + threadIdx.x;   // linear over N*K (dst-major)
    if (i >= K * N) return;
    int k = i % K, n = i / K;
    dst[i] = f2bf(src[(size_t)k * N + n]);
}

// ---------------------------------------------------------------------------
// LayerNorm over D=768, one row per block (256 threads x 3 elements)
// ---------------------------------------------------------------------------
__global__ void k_layernorm_bf16(const float* __restrict__ x,
                                 const float* __restrict__ g,
                                 const float* __restrict__ be,
                                 ushort_t* __restrict__ out) {
    __shared__ float r1[256];
    __shared__ float r2[256];
    int row = blockIdx.x, t = threadIdx.x;
    const float* xr = x + (size_t)row * D_MODEL;
    float a0 = xr[t], a1 = xr[t + 256], a2 = xr[t + 512];
    r1[t] = a0 + a1 + a2;
    r2[t] = a0 * a0 + a1 * a1 + a2 * a2;
    __syncthreads();
    for (int s = 128; s > 0; s >>= 1) {
        if (t < s) { r1[t] += r1[t + s]; r2[t] += r2[t + s]; }
        __syncthreads();
    }
    float mean = r1[0] * (1.0f / D_MODEL);
    float var  = r2[0] * (1.0f / D_MODEL) - mean * mean;
    float inv  = rsqrtf(var + 1e-5f);
    ushort_t* orow = out + (size_t)row * D_MODEL;
    orow[t]       = f2bf((a0 - mean) * inv * g[t]       + be[t]);
    orow[t + 256] = f2bf((a1 - mean) * inv * g[t + 256] + be[t + 256]);
    orow[t + 512] = f2bf((a2 - mean) * inv * g[t + 512] + be[t + 512]);
}

// ---------------------------------------------------------------------------
// WMMA GEMM: out = epilogue(A[M x K]_bf16 @ Wt[N x K]_bf16^T + bias)
// Macro tile 64(M) x 128(N), K-step 32, 8 waves each owning a 32x32 C tile.
// Tile staging via TDM (tensor_load_to_lds) with ping-pong double buffering:
// wave 0 issues next tile's DMA, waits tensorcnt<=2 (current tiles resident,
// next still streaming), and all waves compute from the current buffer.
// MODE 0: bf16 out, scaled by alpha        (QKV projections)
// MODE 1: bf16 out, exact GELU             (FFN up-proj)
// MODE 2: f32 out, + residual              (O-proj, FFN down-proj)
// ---------------------------------------------------------------------------
template <int MODE>
__global__ void k_gemm_wmma(const ushort_t* __restrict__ A,
                            const ushort_t* __restrict__ Wt,
                            const float* __restrict__ bias,
                            const float* __restrict__ resid,
                            void* __restrict__ outp,
                            int M, int N, int K, float alpha) {
    __shared__ ushort_t lds_a[2][64 * 32];
    __shared__ ushort_t lds_b[2][128 * 32];
    int t    = threadIdx.x;
    int lane = t & 31, wave = t >> 5;
    int half = lane >> 4, l16 = lane & 15;
    int waveM = wave >> 2, waveN = wave & 3;
    int mBase = blockIdx.y * 64;
    int nBase = blockIdx.x * 128;

    v8f acc[2][2] = {};
    int nk = K / 32;

#if HAS_TDM
    if (wave == 0) {  // prologue: stage k-step 0 into buffer 0
        tdm_load_2d(A  + (size_t)mBase * K, lds_a[0], 32, 64,  (unsigned)K);
        tdm_load_2d(Wt + (size_t)nBase * K, lds_b[0], 32, 128, (unsigned)K);
    }
#endif

    for (int ki = 0; ki < nk; ki++) {
        int cur = ki & 1;
#if HAS_TDM
        if (wave == 0) {
            if (ki + 1 < nk) {
                int kk2 = (ki + 1) * 32;
                tdm_load_2d(A  + (size_t)mBase * K + kk2, lds_a[cur ^ 1], 32, 64,  (unsigned)K);
                tdm_load_2d(Wt + (size_t)nBase * K + kk2, lds_b[cur ^ 1], 32, 128, (unsigned)K);
                __builtin_amdgcn_s_wait_tensorcnt(2);  // 2 outstanding = next A+B
            } else {
                __builtin_amdgcn_s_wait_tensorcnt(0);
            }
        }
        __syncthreads();   // lds_[ab][cur] resident for all waves
#else
        {
            int kk = ki * 32;
            { // A tile: 64 rows x 32 cols, 1 x b128 per thread
                int row = t >> 2, c = (t & 3) * 8;
                *(uint4*)(lds_a[cur] + row * 32 + c) =
                    *(const uint4*)(A + (size_t)(mBase + row) * K + kk + c);
            }
#pragma unroll
            for (int i = 0; i < 2; i++) { // B tile: 128 rows x 32 cols
                int chunk = t * 2 + i;
                int row = chunk >> 2, c = (chunk & 3) * 8;
                *(uint4*)(lds_b[cur] + row * 32 + c) =
                    *(const uint4*)(Wt + (size_t)(nBase + row) * K + kk + c);
            }
        }
        __syncthreads();
#endif

        FragBF af[2];
#pragma unroll
        for (int ms = 0; ms < 2; ms++) {
            const ushort_t* p = lds_a[cur] + (waveM * 32 + ms * 16 + l16) * 32 + half * 8;
            af[ms].q[0] = *(const uint4*)(p);
            af[ms].q[1] = *(const uint4*)(p + 16);
        }
#pragma unroll
        for (int ns = 0; ns < 2; ns++) {
            FragBF bfr;
            const ushort_t* p = lds_b[cur] + (waveN * 32 + ns * 16 + l16) * 32 + half * 8;
            bfr.q[0] = *(const uint4*)(p);
            bfr.q[1] = *(const uint4*)(p + 16);
#pragma unroll
            for (int ms = 0; ms < 2; ms++) {
                acc[ms][ns] = __builtin_amdgcn_wmma_f32_16x16x32_bf16(
                    false, af[ms].v, false, bfr.v, (short)0, acc[ms][ns], false, false);
            }
        }
        __syncthreads();   // reads of buffer `cur` complete before reuse
    }

#pragma unroll
    for (int ms = 0; ms < 2; ms++) {
#pragma unroll
        for (int ns = 0; ns < 2; ns++) {
            int gn = nBase + waveN * 32 + ns * 16 + l16;
            float bvx = bias[gn];
#pragma unroll
            for (int j = 0; j < 8; j++) {
                int gm = mBase + waveM * 32 + ms * 16 + j + half * 8;
                float v = acc[ms][ns][j] + bvx;
                if (MODE == 0) {
                    ((ushort_t*)outp)[(size_t)gm * N + gn] = f2bf(v * alpha);
                } else if (MODE == 1) {
                    float ge = 0.5f * v * (1.0f + erff(v * 0.70710678118654752f));
                    ((ushort_t*)outp)[(size_t)gm * N + gn] = f2bf(ge);
                } else {
                    ((float*)outp)[(size_t)gm * N + gn] =
                        v + resid[(size_t)gm * N + gn];
                }
            }
        }
    }
}

// ---------------------------------------------------------------------------
// Flash attention: one block per (b, h, 64 q-rows); 4 waves x 16 q-rows.
// kv streamed in blocks of 32; online softmax (VALU permlane reductions);
// all matmuls via WMMA bf16. K tile staged via TDM; V register-transposed.
// Q has 1/sqrt(dk) pre-folded at projection time.
// ---------------------------------------------------------------------------
__global__ void k_flash_attn(const ushort_t* __restrict__ Qb,
                             const ushort_t* __restrict__ Kb,
                             const ushort_t* __restrict__ Vb,
                             const int* __restrict__ amask,
                             ushort_t* __restrict__ ctx) {
    __shared__ ushort_t lds_k[32 * 64];    // K block, row-major [kv][d]
    __shared__ ushort_t lds_vt[64 * 32];   // V block transposed [d][kv]
    __shared__ float    lds_p[4][16 * 32]; // per-wave P scratch [q][kv]
    int t    = threadIdx.x;
    int lane = t & 31, wave = t >> 5;
    int half = lane >> 4, l16 = lane & 15;
    int bh = blockIdx.y;
    int b = bh / N_HEADS, h = bh % N_HEADS;
    int qBase = blockIdx.x * 64 + wave * 16;

    const ushort_t* Qh = Qb + (size_t)b * SEQ * D_MODEL + (size_t)h * DK;
    const ushort_t* Kh = Kb + (size_t)b * SEQ * D_MODEL + (size_t)h * DK;
    const ushort_t* Vh = Vb + (size_t)b * SEQ * D_MODEL + (size_t)h * DK;
    const int* am = amask + b * SEQ;

    // Q fragments (16 q-rows x 64 d), resident for the whole kv loop
    FragBF qf[2];
    {
        int r = qBase + l16;
#pragma unroll
        for (int dc = 0; dc < 2; dc++) {
            const ushort_t* p = Qh + (size_t)r * D_MODEL + dc * 32 + half * 8;
            qf[dc].q[0] = *(const uint4*)p;
            qf[dc].q[1] = *(const uint4*)(p + 16);
        }
    }

    v8f ctxAcc[4] = {};
    float mrun[8], lrun[8];
#pragma unroll
    for (int j = 0; j < 8; j++) { mrun[j] = -3.0e38f; lrun[j] = 0.0f; }

    for (int kb = 0; kb < SEQ / 32; kb++) {
        int kvBase = kb * 32;
        __syncthreads();  // prior iteration's LDS reads complete
#if HAS_TDM
        if (wave == 0)    // K block 32 rows x 64 cols, row stride D_MODEL
            tdm_load_2d(Kh + (size_t)kvBase * D_MODEL, lds_k, 64, 32, D_MODEL);
#else
#pragma unroll
        for (int i = 0; i < 2; i++) { // K block 32x64: 2 x b128 per thread
            int chunk = t * 2 + i;
            int row = chunk >> 3, c = (chunk & 7) * 8;
            *(uint4*)(lds_k + row * 64 + c) =
                *(const uint4*)(Kh + (size_t)(kvBase + row) * D_MODEL + c);
        }
#endif
        { // V block 32x64 -> transposed LDS [d][kv]
            int row = t & 31, seg = t >> 5;
            FragBF tmp;
            const ushort_t* p = Vh + (size_t)(kvBase + row) * D_MODEL + seg * 16;
            tmp.q[0] = *(const uint4*)p;
            tmp.q[1] = *(const uint4*)(p + 8);
#pragma unroll
            for (int e = 0; e < 16; e++)
                lds_vt[(seg * 16 + e) * 32 + row] = tmp.s[e];
        }
#if HAS_TDM
        if (wave == 0) __builtin_amdgcn_s_wait_tensorcnt(0);
#endif
        __syncthreads();

        // scores: 16 q x 32 kv as two 16x16 C tiles, K-dim = dk = 64
        v8f s01[2];
#pragma unroll
        for (int sub = 0; sub < 2; sub++) {
            v8f acc = {};
#pragma unroll
            for (int dc = 0; dc < 2; dc++) {
                FragBF bfr;
                const ushort_t* p = lds_k + (sub * 16 + l16) * 64 + dc * 32 + half * 8;
                bfr.q[0] = *(const uint4*)p;
                bfr.q[1] = *(const uint4*)(p + 16);
                acc = __builtin_amdgcn_wmma_f32_16x16x32_bf16(
                    false, qf[dc].v, false, bfr.v, (short)0, acc, false, false);
            }
            s01[sub] = acc;
        }
        float mb0 = (am[kvBase + l16] == 0)      ? -1.0e30f : 0.0f;
        float mb1 = (am[kvBase + 16 + l16] == 0) ? -1.0e30f : 0.0f;

        // online softmax per q-row (row = j + 8*half; cols live across 16 lanes)
#pragma unroll
        for (int j = 0; j < 8; j++) {
            float s0 = s01[0][j] + mb0;
            float s1 = s01[1][j] + mb1;
            float vmax  = redmax16(fmaxf(s0, s1));
            float newm  = fmaxf(mrun[j], vmax);
            float scale = __expf(mrun[j] - newm);
            mrun[j] = newm;
            float p0 = __expf(s0 - newm);
            float p1 = __expf(s1 - newm);
            float rs = redsum16(p0 + p1);
            lrun[j] = lrun[j] * scale + rs;
#pragma unroll
            for (int tt = 0; tt < 4; tt++) ctxAcc[tt][j] *= scale;
            int prow = j + half * 8;
            lds_p[wave][prow * 32 + l16]      = p0;
            lds_p[wave][prow * 32 + 16 + l16] = p1;
        }
        __syncthreads();

        // P (16x32) as bf16 A fragment via LDS re-layout
        FragBF pf;
        {
            const float* pp = &lds_p[wave][l16 * 32];
#pragma unroll
            for (int e = 0; e < 8; e++) {
                pf.s[e]     = f2bf(pp[half * 8 + e]);
                pf.s[8 + e] = f2bf(pp[16 + half * 8 + e]);
            }
        }
        // ctx += P @ V  (four 16x16 d-tiles)
#pragma unroll
        for (int tt = 0; tt < 4; tt++) {
            FragBF bfr;
            const ushort_t* p = lds_vt + (tt * 16 + l16) * 32 + half * 8;
            bfr.q[0] = *(const uint4*)p;
            bfr.q[1] = *(const uint4*)(p + 16);
            ctxAcc[tt] = __builtin_amdgcn_wmma_f32_16x16x32_bf16(
                false, pf.v, false, bfr.v, (short)0, ctxAcc[tt], false, false);
        }
    }

    // normalize and emit ctx (bf16) back into [b][s][h*64+d] layout
#pragma unroll
    for (int tt = 0; tt < 4; tt++) {
        int gcol = h * DK + tt * 16 + l16;
#pragma unroll
        for (int j = 0; j < 8; j++) {
            int grow = qBase + j + half * 8;
            float l = lrun[j];
            float v = (l > 0.0f) ? ctxAcc[tt][j] / l : 0.0f;
            ctx[((size_t)b * SEQ + grow) * D_MODEL + gcol] = f2bf(v);
        }
    }
}

// ---------------------------------------------------------------------------
extern "C" void kernel_launch(void* const* d_in, const int* in_sizes, int n_in,
                              void* d_out, int out_size, void* d_ws, size_t ws_size,
                              hipStream_t stream) {
    (void)in_sizes; (void)n_in; (void)out_size; (void)ws_size;
    const float* x   = (const float*)d_in[0];
    const int* amask = (const int*)d_in[1];
    const float* Wq  = (const float*)d_in[2];
    const float* bq  = (const float*)d_in[3];
    const float* Wk  = (const float*)d_in[4];
    const float* bk  = (const float*)d_in[5];
    const float* Wv  = (const float*)d_in[6];
    const float* bv  = (const float*)d_in[7];
    const float* Wo  = (const float*)d_in[8];
    const float* bo  = (const float*)d_in[9];
    const float* g1  = (const float*)d_in[10];
    const float* be1 = (const float*)d_in[11];
    const float* g2  = (const float*)d_in[12];
    const float* be2 = (const float*)d_in[13];
    const float* W1  = (const float*)d_in[14];
    const float* b1  = (const float*)d_in[15];
    const float* W2  = (const float*)d_in[16];
    const float* b2  = (const float*)d_in[17];

    char* ws = (char*)d_ws;
    size_t off = 0;
    auto alloc = [&](size_t bytes) -> void* {
        void* p = ws + off;
        off += (bytes + 255) & ~(size_t)255;
        return p;
    };
    ushort_t* wtq  = (ushort_t*)alloc((size_t)D_MODEL * D_MODEL * 2);
    ushort_t* wtk  = (ushort_t*)alloc((size_t)D_MODEL * D_MODEL * 2);
    ushort_t* wtv  = (ushort_t*)alloc((size_t)D_MODEL * D_MODEL * 2);
    ushort_t* wto  = (ushort_t*)alloc((size_t)D_MODEL * D_MODEL * 2);
    ushort_t* wt1  = (ushort_t*)alloc((size_t)D_FF * D_MODEL * 2);
    ushort_t* wt2  = (ushort_t*)alloc((size_t)D_MODEL * D_FF * 2);
    ushort_t* h1   = (ushort_t*)alloc((size_t)MS * D_MODEL * 2);
    ushort_t* qb   = (ushort_t*)alloc((size_t)MS * D_MODEL * 2);
    ushort_t* kbuf = (ushort_t*)alloc((size_t)MS * D_MODEL * 2);
    ushort_t* vbuf = (ushort_t*)alloc((size_t)MS * D_MODEL * 2);
    ushort_t* ctxb = (ushort_t*)alloc((size_t)MS * D_MODEL * 2);
    float*    attn = (float*)   alloc((size_t)MS * D_MODEL * 4);
    ushort_t* h2   = (ushort_t*)alloc((size_t)MS * D_MODEL * 2);
    ushort_t* fbuf = (ushort_t*)alloc((size_t)MS * D_FF * 2);

    // weights -> bf16, transposed to [out][in]
    k_transpose_bf16<<<(D_MODEL * D_MODEL) / 256, 256, 0, stream>>>(Wq, wtq, D_MODEL, D_MODEL);
    k_transpose_bf16<<<(D_MODEL * D_MODEL) / 256, 256, 0, stream>>>(Wk, wtk, D_MODEL, D_MODEL);
    k_transpose_bf16<<<(D_MODEL * D_MODEL) / 256, 256, 0, stream>>>(Wv, wtv, D_MODEL, D_MODEL);
    k_transpose_bf16<<<(D_MODEL * D_MODEL) / 256, 256, 0, stream>>>(Wo, wto, D_MODEL, D_MODEL);
    k_transpose_bf16<<<(D_MODEL * D_FF) / 256, 256, 0, stream>>>(W1, wt1, D_MODEL, D_FF);
    k_transpose_bf16<<<(D_MODEL * D_FF) / 256, 256, 0, stream>>>(W2, wt2, D_FF, D_MODEL);

    // LN1
    k_layernorm_bf16<<<MS, 256, 0, stream>>>(x, g1, be1, h1);

    // QKV projections (Q pre-scaled by 1/sqrt(dk))
    dim3 gD(D_MODEL / 128, MS / 64);
    k_gemm_wmma<0><<<gD, 256, 0, stream>>>(h1, wtq, bq, nullptr, qb,   MS, D_MODEL, D_MODEL, 0.125f);
    k_gemm_wmma<0><<<gD, 256, 0, stream>>>(h1, wtk, bk, nullptr, kbuf, MS, D_MODEL, D_MODEL, 1.0f);
    k_gemm_wmma<0><<<gD, 256, 0, stream>>>(h1, wtv, bv, nullptr, vbuf, MS, D_MODEL, D_MODEL, 1.0f);

    // flash attention
    k_flash_attn<<<dim3(SEQ / 64, BATCH * N_HEADS), 128, 0, stream>>>(qb, kbuf, vbuf, amask, ctxb);

    // O projection + residual
    k_gemm_wmma<2><<<gD, 256, 0, stream>>>(ctxb, wto, bo, x, attn, MS, D_MODEL, D_MODEL, 1.0f);

    // LN2
    k_layernorm_bf16<<<MS, 256, 0, stream>>>(attn, g2, be2, h2);

    // FFN
    k_gemm_wmma<1><<<dim3(D_FF / 128, MS / 64), 256, 0, stream>>>(h2, wt1, b1, nullptr, fbuf, MS, D_FF, D_MODEL, 1.0f);
    k_gemm_wmma<2><<<gD, 256, 0, stream>>>(fbuf, wt2, b2, attn, (float*)d_out, MS, D_MODEL, D_FF, 1.0f);
}
